// MultiHeadAttention_34454227648681
// MI455X (gfx1250) — matrix operand order repaired and checked by reading, not compiled, not run on hardware
//
#include <hip/hip_runtime.h>
#include <stdint.h>

// ---------------------------------------------------------------------------
// MI455X (gfx1250) multi-head attention, bf16 WMMA pipeline, f32 accumulate.
//   B=2, S=2048, D=1024, H=16, hd=64
//   GEMMs stage LDS tiles with GLOBAL_LOAD_ASYNC_TO_LDS_B128 (double-buffered).
// ---------------------------------------------------------------------------

#define SEQ    2048
#define DMODEL 1024
#define HEADS  16
#define HD     64
#define BATCH  2
#define MROWS  (BATCH * SEQ)        // 4096
#define N_QKV  (3 * DMODEL)         // 3072

typedef __attribute__((ext_vector_type(16))) __bf16 v16bf;
typedef __attribute__((ext_vector_type(8)))  float  v8f;

union FragBF {
    v16bf m;
    uint4 q[2];
};

__device__ __forceinline__ unsigned short f32_to_bf16(float f) {
    union { float f; unsigned u; } v; v.f = f;
    unsigned r = v.u + 0x7FFFu + ((v.u >> 16) & 1u);   // round-to-nearest-even
    return (unsigned short)(r >> 16);
}

__device__ __forceinline__ v8f v8f_zero() {
    v8f v;
#pragma unroll
    for (int i = 0; i < 8; ++i) v[i] = 0.0f;
    return v;
}

__device__ __forceinline__ v8f wmma_bf16(const FragBF& a, const FragBF& b, v8f c) {
    // D = A(16x32 bf16) * B(32x16 bf16) + C(16x16 f32)
    return __builtin_amdgcn_wmma_f32_16x16x32_bf16(
        /*neg_a=*/false, a.m, /*neg_b=*/false, b.m,
        /*c_mod=*/(short)0, c, /*reuse_a=*/false, /*reuse_b=*/false);
}

// A-matrix fragment (16x32, 16-bit): lane half selects K base 0/8; VGPRs 4-7 = K+16.
__device__ __forceinline__ void load_frag_a(FragBF& f, const unsigned short* rowp, int half8) {
    f.q[0] = *reinterpret_cast<const uint4*>(rowp + half8);
    f.q[1] = *reinterpret_cast<const uint4*>(rowp + 16 + half8);
}
// B-matrix fragment (32x16, 16-bit): lane half selects K base 0/8; VGPRs 4-7 = K+16.
__device__ __forceinline__ void load_frag_b(FragBF& f, const unsigned short* colp, int koff) {
    f.q[0] = *reinterpret_cast<const uint4*>(colp + koff / 2);
    f.q[1] = *reinterpret_cast<const uint4*>(colp + 16 + koff / 2);
}

// one b128 per lane: global -> LDS via the async engine (ASYNCcnt-tracked)
__device__ __forceinline__ void async_b128(unsigned lds_off, const void* gaddr) {
    asm volatile("global_load_async_to_lds_b128 %0, %1, off"
                 :: "v"(lds_off), "v"(gaddr) : "memory");
}
__device__ __forceinline__ void wait_async0() {
    asm volatile("s_wait_asynccnt 0x0" ::: "memory");
}

// ---------------------------------------------------------------------------
// Kernel 1a: f32 -> bf16 conversion (x, row-major kept)
// ---------------------------------------------------------------------------
__global__ __launch_bounds__(256) void cvt_bf16_kernel(
    const float* __restrict__ in, unsigned short* __restrict__ out) {
    size_t i = ((size_t)blockIdx.x * 256 + threadIdx.x) * 4;
    float4 v = *reinterpret_cast<const float4*>(in + i);
    uint2 p;
    p.x = (unsigned)f32_to_bf16(v.x) | ((unsigned)f32_to_bf16(v.y) << 16);
    p.y = (unsigned)f32_to_bf16(v.z) | ((unsigned)f32_to_bf16(v.w) << 16);
    *reinterpret_cast<uint2*>(out + i) = p;
}

// ---------------------------------------------------------------------------
// Kernel 1b: one-time weight transpose+convert  W[K][N] f32 -> Wt[N][K] bf16.
// Amortized over the 64 M-tile blocks that reuse each weight tile.
// ---------------------------------------------------------------------------
__global__ __launch_bounds__(256) void transpose_cvt_kernel(
    const float* __restrict__ W, unsigned short* __restrict__ Wt, int K, int N) {
    const int kt = blockIdx.y * 32, nt = blockIdx.x * 32;
    const int r  = threadIdx.x >> 3;         // k within tile, 0..31
    const int c4 = (threadIdx.x & 7) * 4;    // n within tile, 0..28
    float4 v = *reinterpret_cast<const float4*>(W + (size_t)(kt + r) * N + nt + c4);
    Wt[(size_t)(nt + c4 + 0) * K + kt + r] = f32_to_bf16(v.x);
    Wt[(size_t)(nt + c4 + 1) * K + kt + r] = f32_to_bf16(v.y);
    Wt[(size_t)(nt + c4 + 2) * K + kt + r] = f32_to_bf16(v.z);
    Wt[(size_t)(nt + c4 + 3) * K + kt + r] = f32_to_bf16(v.w);
}

// ---------------------------------------------------------------------------
// Kernel 2/4: tiled GEMM, C[M,N] = A_bf16[M,K] * Bt_bf16[N,K]^T + bias[N]
//   Block tile 64x64, BK=32, 256 threads = 8 waves, each wave -> 16x32 of C.
//   Both LDS tiles staged by GLOBAL_LOAD_ASYNC_TO_LDS_B128, double-buffered:
//   issue tile k+1, compute tile k, s_wait_asynccnt 0, barrier.
//   MODE 0: scatter epilogue into per-head Q, K ([s][hd]) and V^T ([hd][s]).
//   MODE 1: plain f32 store (output projection).
// ---------------------------------------------------------------------------
#define LDT 40   // padded row stride (bf16 elems), 80B: 16B aligned, bank-spread

template <int MODE>
__global__ __launch_bounds__(256) void gemm_bf16_kernel(
    const unsigned short* __restrict__ A,    // [M x K] bf16 row-major
    const unsigned short* __restrict__ Bt,   // [N x K] bf16 row-major (W^T)
    const float* __restrict__ bias,          // [N]
    float* __restrict__ Cout,                // MODE 1
    unsigned short* __restrict__ q_ws,       // MODE 0
    unsigned short* __restrict__ k_ws,       // MODE 0
    unsigned short* __restrict__ vT_ws,      // MODE 0
    int M, int N, int K) {

    __shared__ __align__(16) unsigned short lA[2][64 * LDT];   // A tile  [row][k]
    __shared__ __align__(16) unsigned short lB[2][64 * LDT];   // Bt tile [n][k]

    const int tid   = threadIdx.x;
    const int lane  = tid & 31;
    const int wave  = tid >> 5;          // 0..7
    const int wm    = wave >> 1;         // 0..3 : 16-row strip
    const int wn    = wave & 1;          // 0..1 : 32-col strip
    const int m0    = blockIdx.y * 64;
    const int n0    = blockIdx.x * 64;
    const int lc    = lane & 15;
    const int half8 = (lane & 16) ? 8 : 0;
    const int koff  = (lane & 16) ? 16 : 0;

    // per-thread staging slot: 64 rows x 4 chunks of 8 bf16 (16B)
    const int srow = tid >> 2, sch = tid & 3;
    const unsigned short* gA = A  + (size_t)(m0 + srow) * K + sch * 8;
    const unsigned short* gB = Bt + (size_t)(n0 + srow) * K + sch * 8;
    const unsigned aoff[2] = {
        (unsigned)(uintptr_t)&lA[0][srow * LDT + sch * 8],
        (unsigned)(uintptr_t)&lA[1][srow * LDT + sch * 8] };
    const unsigned boff[2] = {
        (unsigned)(uintptr_t)&lB[0][srow * LDT + sch * 8],
        (unsigned)(uintptr_t)&lB[1][srow * LDT + sch * 8] };

    v8f acc0 = v8f_zero();
    v8f acc1 = v8f_zero();

    // prologue: stage tile 0
    async_b128(aoff[0], gA);
    async_b128(boff[0], gB);
    wait_async0();
    __syncthreads();

    const int nsteps = K / 32;
    for (int step = 0; step < nsteps; ++step) {
        const int cur = step & 1;
        if (step + 1 < nsteps) {            // overlap: fetch next while computing
            const int k1 = (step + 1) * 32;
            async_b128(aoff[cur ^ 1], gA + k1);
            async_b128(boff[cur ^ 1], gB + k1);
        }

        FragBF a, b0, b1;
        load_frag_a(a,  &lA[cur][(wm * 16 + lc) * LDT], half8);
        load_frag_b(b0, &lB[cur][(wn * 32 + lc) * LDT], koff);
        load_frag_b(b1, &lB[cur][(wn * 32 + 16 + lc) * LDT], koff);
        acc0 = wmma_bf16(a, b0, acc0);
        acc1 = wmma_bf16(a, b1, acc1);

        wait_async0();                      // next buffer fully in LDS
        __syncthreads();                    // all waves done reading cur
    }

    // ---- epilogue: C/D layout -> row = j + half8, col = lane&15
#pragma unroll
    for (int sn = 0; sn < 2; ++sn) {
        const v8f& c = sn ? acc1 : acc0;
        const int col = n0 + wn * 32 + sn * 16 + lc;
        const float bv = bias[col];
#pragma unroll
        for (int j = 0; j < 8; ++j) {
            const int row = m0 + wm * 16 + j + half8;
            const float val = c[j] + bv;
            if (MODE == 1) {
                Cout[(size_t)row * N + col] = val;
            } else {
                const int b = row >> 11, s = row & (SEQ - 1);
                const int h = col / (3 * HD), r = col % (3 * HD);
                const int bh = b * HEADS + h;
                const unsigned short bw = f32_to_bf16(val);
                if (r < HD)
                    q_ws[((size_t)bh * SEQ + s) * HD + r] = bw;
                else if (r < 2 * HD)
                    k_ws[((size_t)bh * SEQ + s) * HD + (r - HD)] = bw;
                else
                    vT_ws[((size_t)bh * HD + (r - 2 * HD)) * SEQ + s] = bw;
            }
        }
    }
}

// ---------------------------------------------------------------------------
// Kernel 3: causal flash attention. 4 waves/block; each wave owns 16 query
// rows x hd=64, loops over 32-key tiles up to the diagonal. Both matmuls are
// v_wmma_f32_16x16x32_bf16; softmax state (m,l) is per-row f32 maintained via
// 16-lane shfl_xor reductions (rows live in a 16-lane half per C/D layout).
// ---------------------------------------------------------------------------
__global__ __launch_bounds__(128) void attn_kernel(
    const unsigned short* __restrict__ q_ws,   // [BH][S][hd] bf16
    const unsigned short* __restrict__ k_ws,   // [BH][S][hd] bf16
    const unsigned short* __restrict__ vT_ws,  // [BH][hd][S] bf16
    unsigned short* __restrict__ ctx_ws) {     // [B][S][D]  bf16

    const int bh    = blockIdx.x;            // 0..31 (b*16+h)
    const int tid   = threadIdx.x;
    const int lane  = tid & 31;
    const int wave  = tid >> 5;              // 0..3
    const int qbase = blockIdx.y * 64 + wave * 16;
    const int lc    = lane & 15;
    const int half8 = (lane & 16) ? 8 : 0;
    const int koff  = (lane & 16) ? 16 : 0;
    const float NEG_INF = -__builtin_inff();

    __shared__ __align__(16) unsigned short ldsP[4][16 * LDT];  // per-wave P tile
    unsigned short* myP = ldsP[wave];

    // Q fragments (A-matrix), d-chunks 0..31 and 32..63; loaded once.
    FragBF qf[2];
    {
        const unsigned short* pq = q_ws + ((size_t)bh * SEQ + qbase + lc) * HD;
        load_frag_a(qf[0], pq, half8);
        load_frag_a(qf[1], pq + 32, half8);
    }

    v8f o[4];
#pragma unroll
    for (int dn = 0; dn < 4; ++dn) o[dn] = v8f_zero();
    float mrow[8], lrow[8];
#pragma unroll
    for (int j = 0; j < 8; ++j) { mrow[j] = NEG_INF; lrow[j] = 0.0f; }

    const int ktend = (qbase + 15) >> 5;     // last key tile touching diagonal
    for (int kt = 0; kt <= ktend; ++kt) {
        const int kbase = kt * 32;

        // prefetch next iteration's K rows / V rows (global_prefetch_b8)
        if (kt < ktend) {
            __builtin_prefetch(k_ws + ((size_t)bh * SEQ + kbase + 32 + lane) * HD, 0, 3);
            __builtin_prefetch(vT_ws + ((size_t)bh * HD + lane) * SEQ + kbase + 32, 0, 3);
        }

        // ---- scores tile: S(16q x 32k) = Q(16 x 64) * K^T(64 x 32)
        v8f s0 = v8f_zero(), s1 = v8f_zero();
#pragma unroll
        for (int dstep = 0; dstep < 2; ++dstep) {
            FragBF kb0, kb1;
            const unsigned short* pk =
                k_ws + ((size_t)bh * SEQ + kbase + lc) * HD + dstep * 32;
            load_frag_b(kb0, pk, koff);
            load_frag_b(kb1, pk + 16 * HD, koff);
            s0 = wmma_bf16(qf[dstep], kb0, s0);
            s1 = wmma_bf16(qf[dstep], kb1, s1);
        }

        // ---- scale, causal mask, online softmax
#pragma unroll
        for (int j = 0; j < 8; ++j) {
            const int qrow = qbase + j + half8;
            float v0 = s0[j] * 0.125f;                 // 1/sqrt(64)
            float v1 = s1[j] * 0.125f;
            if (kbase + lc      > qrow) v0 = NEG_INF;
            if (kbase + 16 + lc > qrow) v1 = NEG_INF;
            float mx = fmaxf(v0, v1);
#pragma unroll
            for (int off = 1; off < 16; off <<= 1)
                mx = fmaxf(mx, __shfl_xor(mx, off, 32));
            const float mn  = fmaxf(mrow[j], mx);
            const float scl = __expf(mrow[j] - mn);    // first iter: exp(-inf)=0
            const float p0  = __expf(v0 - mn);
            const float p1  = __expf(v1 - mn);
            float rs = p0 + p1;
#pragma unroll
            for (int off = 1; off < 16; off <<= 1)
                rs += __shfl_xor(rs, off, 32);
            lrow[j] = lrow[j] * scl + rs;
            mrow[j] = mn;
#pragma unroll
            for (int dn = 0; dn < 4; ++dn) o[dn][j] *= scl;
            // P (C-layout) -> LDS bf16 for re-read in A-layout
            myP[(j + half8) * LDT + lc]      = f32_to_bf16(p0);
            myP[(j + half8) * LDT + 16 + lc] = f32_to_bf16(p1);
        }

        // cross-lane DS store -> load hazard within the wave
        asm volatile("s_wait_dscnt 0x0" ::: "memory");

        FragBF pf;
        load_frag_a(pf, &myP[lc * LDT], half8);

        // ---- O += P(16 x 32) * V(32 x 64), 4 column sub-tiles from V^T
#pragma unroll
        for (int dn = 0; dn < 4; ++dn) {
            FragBF vb;
            const unsigned short* pv =
                vT_ws + ((size_t)bh * HD + dn * 16 + lc) * SEQ + kbase;
            load_frag_b(vb, pv, koff);
            o[dn] = wmma_bf16(pf, vb, o[dn]);
        }
    }

    // ---- finalize: O /= l, store ctx as bf16 [b][s][h*64 + d]
    const int b = bh >> 4, h = bh & 15;
#pragma unroll
    for (int j = 0; j < 8; ++j) {
        const float inv  = 1.0f / lrow[j];
        const int   qrow = qbase + j + half8;
        const size_t base =
            ((size_t)b * SEQ + qrow) * DMODEL + h * HD + lc;
#pragma unroll
        for (int dn = 0; dn < 4; ++dn)
            ctx_ws[base + dn * 16] = f32_to_bf16(o[dn][j] * inv);
    }
}

// ---------------------------------------------------------------------------
// Host-side launch
// ---------------------------------------------------------------------------
extern "C" void kernel_launch(void* const* d_in, const int* in_sizes, int n_in,
                              void* d_out, int out_size, void* d_ws, size_t ws_size,
                              hipStream_t stream) {
    const float* x    = (const float*)d_in[0];
    const float* Wqkv = (const float*)d_in[1];
    const float* bqkv = (const float*)d_in[2];
    const float* Wo   = (const float*)d_in[3];
    const float* bo   = (const float*)d_in[4];
    float* out = (float*)d_out;

    // workspace carve-up
    const size_t SEG = (size_t)MROWS * DMODEL * sizeof(unsigned short); // 8 MiB
    char* ws = (char*)d_ws;
    unsigned short* x_bf   = (unsigned short*)(ws + 0 * SEG);
    unsigned short* q_ws   = (unsigned short*)(ws + 1 * SEG);
    unsigned short* k_ws   = (unsigned short*)(ws + 2 * SEG);
    unsigned short* vT_ws  = (unsigned short*)(ws + 3 * SEG);
    unsigned short* ctx_ws = (unsigned short*)(ws + 4 * SEG);
    unsigned short* WqkvT  = (unsigned short*)(ws + 5 * SEG);               // 6 MiB
    unsigned short* WoT    = (unsigned short*)((char*)WqkvT +
                              (size_t)N_QKV * DMODEL * sizeof(unsigned short)); // 2 MiB

    // 1) x -> bf16 ; weights -> transposed bf16 [N][K]
    cvt_bf16_kernel<<<(MROWS * DMODEL) / (256 * 4), 256, 0, stream>>>(x, x_bf);
    transpose_cvt_kernel<<<dim3(N_QKV / 32, DMODEL / 32), 256, 0, stream>>>(
        Wqkv, WqkvT, DMODEL, N_QKV);
    transpose_cvt_kernel<<<dim3(DMODEL / 32, DMODEL / 32), 256, 0, stream>>>(
        Wo, WoT, DMODEL, DMODEL);

    // 2) QKV projection + head split (M=4096, N=3072, K=1024)
    gemm_bf16_kernel<0><<<dim3(N_QKV / 64, MROWS / 64), 256, 0, stream>>>(
        x_bf, WqkvT, bqkv, nullptr, q_ws, k_ws, vT_ws, MROWS, N_QKV, DMODEL);

    // 3) causal flash attention: grid (B*H, S/64), 4 waves each
    attn_kernel<<<dim3(BATCH * HEADS, SEQ / 64), 128, 0, stream>>>(
        q_ws, k_ws, vT_ws, ctx_ws);

    // 4) output projection (M=4096, N=1024, K=1024)
    gemm_bf16_kernel<1><<<dim3(DMODEL / 64, MROWS / 64), 256, 0, stream>>>(
        ctx_ws, WoT, bo, out, nullptr, nullptr, nullptr, MROWS, DMODEL, DMODEL);
}
